// MultiHeadAttention_61572651155560
// MI455X (gfx1250) — matrix operand order repaired and checked by reading, not compiled, not run on hardware
//
#include <hip/hip_runtime.h>
#include <hip/hip_bf16.h>

// ---------- types ----------
typedef __attribute__((ext_vector_type(16))) _Float16 v16h;
typedef __attribute__((ext_vector_type(8)))  _Float16 v8h;
typedef __attribute__((ext_vector_type(8)))  float    v8f;

static __device__ __forceinline__ v16h cat8(v8h lo, v8h hi) {
  return __builtin_shufflevector(lo, hi, 0,1,2,3,4,5,6,7,8,9,10,11,12,13,14,15);
}

// DPP butterfly reduction over each 16-lane half (rows of the WMMA C layout)
template <int CTRL>
static __device__ __forceinline__ float dpp_f32(float x) {
  return __int_as_float(
      __builtin_amdgcn_update_dpp(0, __float_as_int(x), CTRL, 0xF, 0xF, true));
}
static __device__ __forceinline__ float red16_max(float x) {
  x = fmaxf(x, dpp_f32<0xB1>(x));   // quad_perm xor1
  x = fmaxf(x, dpp_f32<0x4E>(x));   // quad_perm xor2
  x = fmaxf(x, dpp_f32<0x141>(x));  // row_half_mirror
  x = fmaxf(x, dpp_f32<0x140>(x));  // row_mirror
  return x;
}
static __device__ __forceinline__ float red16_sum(float x) {
  x += dpp_f32<0xB1>(x);
  x += dpp_f32<0x4E>(x);
  x += dpp_f32<0x141>(x);
  x += dpp_f32<0x140>(x);
  return x;
}

// async copy: 16 bytes global -> LDS, tracked on ASYNCcnt
static __device__ __forceinline__ void async_b128(const _Float16* gsrc, _Float16* ldst) {
  unsigned l = (unsigned)(size_t)ldst;                 // low 32 bits = LDS offset
  unsigned long long g = (unsigned long long)(size_t)gsrc;
  asm volatile("global_load_async_to_lds_b128 %0, %1, off" :: "v"(l), "v"(g) : "memory");
}
static __device__ __forceinline__ void wait_async0() {
  asm volatile("s_wait_asynccnt 0" ::: "memory");
}

// Problem constants
#define BQ   2
#define SQ   2048
#define DIN  1024
#define HH   16
#define DHD  64
#define DMOD 1024
#define DOUT 512
#define MROWS (BQ*SQ)          // 4096
#define NQKV  (3*DMOD)         // 3072

// ---------- cast kernels ----------
__global__ void cast_f32_f16(const float* __restrict__ src, _Float16* __restrict__ dst, int n) {
  int i = blockIdx.x * 256 + threadIdx.x;
  if (i < n) dst[i] = (_Float16)src[i];
}

__global__ void pack_wqkv(const float* __restrict__ Wq, const float* __restrict__ Wk,
                          const float* __restrict__ Wv, _Float16* __restrict__ dst) {
  int c = blockIdx.x * 256 + threadIdx.x;   // 0..3071
  int k = blockIdx.y;                       // 0..1023
  int m = c >> 10, e = c & 1023;
  const float* W = (m == 0) ? Wq : ((m == 1) ? Wk : Wv);
  dst[(size_t)k * NQKV + c] = (_Float16)W[(size_t)k * DMOD + e];
}

// ---------- tiled f16 WMMA GEMM, double-buffered with async A staging ----------
// C[M,N] = A[M,K] * B[K,N] + bias ; EPI==0: scatter to Q/K/V f16 (Q scaled 1/8)
//                                   EPI==1: f32 row-major out
#define LDS_STRIDE 40  // 32 + 8 pad halves

template <int EPI>
__global__ __launch_bounds__(256)
void gemm16(const _Float16* __restrict__ A, const _Float16* __restrict__ Bw,
            const float* __restrict__ bias,
            _Float16* __restrict__ outH, float* __restrict__ outF,
            int Md, int Nd, int Kd) {
  __shared__ _Float16 sA [2][128 * LDS_STRIDE];
  __shared__ _Float16 sBt[2][128 * LDS_STRIDE];

  const int tid  = threadIdx.x;
  const int wave = tid >> 5, lane = tid & 31;
  const int wm = wave >> 1, wn = wave & 1;       // 4 x 2 wave grid
  const int bm = blockIdx.y * 128, bn = blockIdx.x * 128;
  const int l15 = lane & 15;
  const int hi  = (lane >> 4) & 1;

  v8f acc[2][4];
#pragma unroll
  for (int i = 0; i < 2; i++)
#pragma unroll
    for (int j = 0; j < 4; j++) { v8f z = {}; acc[i][j] = z; }

  // stage tile at k0 into buffer buf
  auto stage = [&](int k0, int buf) {
    // A tile 128x32 row-major: async global->LDS, no VGPR round trip
    int row = tid >> 2;          // 0..63
    int kc  = (tid & 3) * 8;     // 0,8,16,24
#pragma unroll
    for (int p = 0; p < 2; p++) {
      int r = row + p * 64;
      async_b128(A + (size_t)(bm + r) * Kd + k0 + kc,
                 &sA[buf][r * LDS_STRIDE + kc]);
    }
    // B tile 32x128 transposed into sBt[n][k] (scatter needs VGPRs)
    int n0 = (tid & 15) * 8;     // 0..120
    int kk = tid >> 4;           // 0..15
#pragma unroll
    for (int p = 0; p < 2; p++) {
      int k = kk + p * 16;
      v8h v = *(const v8h*)(Bw + (size_t)(k0 + k) * Nd + bn + n0);
#pragma unroll
      for (int j = 0; j < 8; j++) sBt[buf][(n0 + j) * LDS_STRIDE + k] = v[j];
    }
  };

  const int nT = Kd >> 5;
  stage(0, 0);

  for (int t = 0; t < nT; t++) {
    const int cur = t & 1;
    wait_async0();        // our async stores into buf 'cur' are done
    __syncthreads();      // compiler adds s_wait_dscnt 0 before barrier
    if (t + 1 < nT) stage((t + 1) << 5, cur ^ 1);

    // fragments from buffer 'cur'
    v16h af[2], bf[4];
#pragma unroll
    for (int tm = 0; tm < 2; tm++) {
      const _Float16* p = &sA[cur][(wm * 32 + tm * 16 + l15) * LDS_STRIDE + hi * 8];
      af[tm] = cat8(*(const v8h*)p, *(const v8h*)(p + 16));
    }
#pragma unroll
    for (int tn = 0; tn < 4; tn++) {
      const _Float16* p = &sBt[cur][(wn * 64 + tn * 16 + l15) * LDS_STRIDE + hi * 8];
      bf[tn] = cat8(*(const v8h*)p, *(const v8h*)(p + 16));
    }
#pragma unroll
    for (int tm = 0; tm < 2; tm++)
#pragma unroll
      for (int tn = 0; tn < 4; tn++)
        acc[tm][tn] = __builtin_amdgcn_wmma_f32_16x16x32_f16(
            false, af[tm], false, bf[tn], (short)0, acc[tm][tn], false, false);
  }

  // --- epilogue ---
#pragma unroll
  for (int tm = 0; tm < 2; tm++) {
#pragma unroll
    for (int tn = 0; tn < 4; tn++) {
      int gn = bn + wn * 64 + tn * 16 + l15;
      float bvl = bias[gn];
      if (EPI == 0) {
        int mtx = gn >> 10;           // 0=Q,1=K,2=V (uniform per 16-tile)
        int em  = gn & 1023;
        int h = em >> 6, dh = em & 63;
        float scl = (mtx == 0) ? 0.125f : 1.0f;   // 1/sqrt(64) folded into Q
#pragma unroll
        for (int r = 0; r < 8; r++) {
          int gm = bm + wm * 32 + tm * 16 + r + hi * 8;
          int b = gm >> 11, s = gm & 2047;
          float v = (acc[tm][tn][r] + bvl) * scl;
          outH[(size_t)mtx * ((size_t)MROWS * DMOD) +
               (((size_t)(b * HH + h)) * SQ + s) * DHD + dh] = (_Float16)v;
        }
      } else {
#pragma unroll
        for (int r = 0; r < 8; r++) {
          int gm = bm + wm * 32 + tm * 16 + r + hi * 8;
          outF[(size_t)gm * Nd + gn] = acc[tm][tn][r] + bvl;
        }
      }
    }
  }
}

// ---------- flash attention: one block per (bh, 128-query tile), 8 waves ----------
#define LKT 72  // 64 + 8 pad halves
__global__ __launch_bounds__(256)
void attn64(const _Float16* __restrict__ Qh, const _Float16* __restrict__ Kh,
            const _Float16* __restrict__ Vh, _Float16* __restrict__ outH) {
  __shared__ _Float16 sK [2][64 * LKT];      // [key][dh]
  __shared__ _Float16 sVt[2][64 * LKT];      // [dh][key]
  __shared__ _Float16 sP [8][16 * LKT];      // per-wave P staging [row][key]

  const int tid  = threadIdx.x;
  const int wave = tid >> 5, lane = tid & 31;
  const int l15 = lane & 15, hi = (lane >> 4) & 1;
  const int bh = blockIdx.y;              // b*16 + h
  const int q0 = blockIdx.x * 128;
  const size_t base = (size_t)bh * SQ * DHD;

  // Q fragments for this wave's 16 rows (pre-scaled by 1/8 in gemm16<0>)
  v16h qa[2];
  {
    const _Float16* qp = Qh + base + (size_t)(q0 + wave * 16 + l15) * DHD;
#pragma unroll
    for (int s = 0; s < 2; s++) {
      const _Float16* p = qp + s * 32 + hi * 8;
      qa[s] = cat8(*(const v8h*)p, *(const v8h*)(p + 16));
    }
  }

  v8f o[4];
#pragma unroll
  for (int i = 0; i < 4; i++) { v8f z = {}; o[i] = z; }
  float mr[8], lr[8];
#pragma unroll
  for (int r = 0; r < 8; r++) { mr[r] = -3.0e30f; lr[r] = 0.0f; }

  // stage 64-key tile kt into buffer buf (async K copy, manual V transpose)
  auto stage = [&](int kt, int buf) {
    const _Float16* srcK = Kh + base + (size_t)kt * 64 * DHD;
    const _Float16* srcV = Vh + base + (size_t)kt * 64 * DHD;
#pragma unroll
    for (int it = 0; it < 2; it++) {
      int item = tid + it * 256;          // 0..511
      int key = item >> 3, d0 = (item & 7) * 8;
      async_b128(srcK + key * DHD + d0, &sK[buf][key * LKT + d0]);
      v8h vv = *(const v8h*)(srcV + key * DHD + d0);
#pragma unroll
      for (int j = 0; j < 8; j++) sVt[buf][(d0 + j) * LKT + key] = vv[j];
    }
  };

  const int nT = SQ / 64;
  stage(0, 0);

  for (int kt = 0; kt < nT; kt++) {
    const int cur = kt & 1;
    wait_async0();
    __syncthreads();
    if (kt + 1 < nT) stage(kt + 1, cur ^ 1);

    // scores S = Q * K^T  (16 rows x 64 keys)
    v8f sc[4];
#pragma unroll
    for (int t = 0; t < 4; t++) {
      v8f c = {};
#pragma unroll
      for (int s = 0; s < 2; s++) {
        const _Float16* p = &sK[cur][(t * 16 + l15) * LKT + s * 32 + hi * 8];
        v16h bfr = cat8(*(const v8h*)p, *(const v8h*)(p + 16));
        c = __builtin_amdgcn_wmma_f32_16x16x32_f16(false, qa[s], false, bfr,
                                                   (short)0, c, false, false);
      }
      sc[t] = c;
    }

    // online softmax: row reductions via DPP butterflies (no LDS traffic)
    float alpha[8];
#pragma unroll
    for (int r = 0; r < 8; r++) {
      float mx = fmaxf(fmaxf(sc[0][r], sc[1][r]), fmaxf(sc[2][r], sc[3][r]));
      mx = red16_max(mx);
      float mn = fmaxf(mr[r], mx);
      alpha[r] = __expf(mr[r] - mn);
      mr[r] = mn;
      float rs = 0.0f;
#pragma unroll
      for (int t = 0; t < 4; t++) {
        float p = __expf(sc[t][r] - mn);
        sc[t][r] = p;
        rs += p;
      }
      lr[r] = lr[r] * alpha[r] + red16_sum(rs);
    }

    // P (C-layout) -> per-wave LDS -> A-layout fragments
    _Float16* pw = sP[wave];
#pragma unroll
    for (int t = 0; t < 4; t++)
#pragma unroll
      for (int r = 0; r < 8; r++)
        pw[(r + hi * 8) * LKT + t * 16 + l15] = (_Float16)sc[t][r];

    v16h pa[2];
    {
      const _Float16* pr = sP[wave] + l15 * LKT;
#pragma unroll
      for (int s = 0; s < 2; s++) {
        const _Float16* p = pr + s * 32 + hi * 8;
        pa[s] = cat8(*(const v8h*)p, *(const v8h*)(p + 16));
      }
    }

    // O = alpha*O + P*V
#pragma unroll
    for (int td = 0; td < 4; td++) {
#pragma unroll
      for (int r = 0; r < 8; r++) o[td][r] *= alpha[r];
#pragma unroll
      for (int s = 0; s < 2; s++) {
        const _Float16* p = &sVt[cur][(td * 16 + l15) * LKT + s * 32 + hi * 8];
        v16h vf = cat8(*(const v8h*)p, *(const v8h*)(p + 16));
        o[td] = __builtin_amdgcn_wmma_f32_16x16x32_f16(false, pa[s], false, vf,
                                                       (short)0, o[td], false, false);
      }
    }
  }

  // normalize + write attn in [b, s, h*64+dh] f16 layout
  const int b = bh >> 4, h = bh & 15;
#pragma unroll
  for (int td = 0; td < 4; td++) {
#pragma unroll
    for (int r = 0; r < 8; r++) {
      int s = q0 + wave * 16 + r + hi * 8;
      int col = h * DHD + td * 16 + l15;
      outH[((size_t)(b * SQ + s)) * DMOD + col] = (_Float16)(o[td][r] / lr[r]);
    }
  }
}

// ---------- workspace layout (bytes) ----------
#define OFF_XH    0ull                      // 4096*1024 f16 = 8 MB
#define OFF_WQKV  8388608ull                // 1024*3072 f16 = 6 MB
#define OFF_WOH   14680064ull               // 1024*512  f16 = 1 MB
#define OFF_BQKV  15728640ull               // 3072 f32
#define OFF_QKV   15740928ull               // 3 * 4096*1024 f16 = 24 MB
#define OFF_ATT   40906752ull               // 4096*1024 f16 = 8 MB

extern "C" void kernel_launch(void* const* d_in, const int* in_sizes, int n_in,
                              void* d_out, int out_size, void* d_ws, size_t ws_size,
                              hipStream_t stream) {
  const float* x  = (const float*)d_in[0];
  const float* Wq = (const float*)d_in[1];
  const float* bq = (const float*)d_in[2];
  const float* Wk = (const float*)d_in[3];
  const float* bk = (const float*)d_in[4];
  const float* Wv = (const float*)d_in[5];
  const float* bv = (const float*)d_in[6];
  const float* Wo = (const float*)d_in[7];
  const float* bo = (const float*)d_in[8];

  char* ws = (char*)d_ws;
  _Float16* Xh   = (_Float16*)(ws + OFF_XH);
  _Float16* Wqkv = (_Float16*)(ws + OFF_WQKV);
  _Float16* WoH  = (_Float16*)(ws + OFF_WOH);
  float*    bqkv = (float*)   (ws + OFF_BQKV);
  _Float16* QKVh = (_Float16*)(ws + OFF_QKV);
  _Float16* AttH = (_Float16*)(ws + OFF_ATT);

  _Float16* Qh = QKVh;
  _Float16* Kh = QKVh + (size_t)MROWS * DMOD;
  _Float16* Vh = QKVh + 2ull * MROWS * DMOD;

  // concat biases (device-to-device async copies are graph-capture safe)
  hipMemcpyAsync(bqkv,        bq, DMOD * sizeof(float), hipMemcpyDeviceToDevice, stream);
  hipMemcpyAsync(bqkv + 1024, bk, DMOD * sizeof(float), hipMemcpyDeviceToDevice, stream);
  hipMemcpyAsync(bqkv + 2048, bv, DMOD * sizeof(float), hipMemcpyDeviceToDevice, stream);

  // casts / packs
  cast_f32_f16<<<(MROWS * DIN + 255) / 256, 256, 0, stream>>>(x, Xh, MROWS * DIN);
  pack_wqkv<<<dim3(NQKV / 256, DIN), 256, 0, stream>>>(Wq, Wk, Wv, Wqkv);
  cast_f32_f16<<<(DMOD * DOUT + 255) / 256, 256, 0, stream>>>(Wo, WoH, DMOD * DOUT);

  // QKV projection: [4096x1024] * [1024x3072], scatter Q/K/V (Q pre-scaled)
  gemm16<0><<<dim3(NQKV / 128, MROWS / 128), 256, 0, stream>>>(
      Xh, Wqkv, bqkv, QKVh, nullptr, MROWS, NQKV, DIN);

  // flash attention per (head, 128-query tile)
  attn64<<<dim3(SQ / 128, BQ * HH), 256, 0, stream>>>(Qh, Kh, Vh, AttH);

  // output projection: [4096x1024] * [1024x512] -> f32 d_out
  gemm16<1><<<dim3(DOUT / 128, MROWS / 128), 256, 0, stream>>>(
      AttH, WoH, bo, nullptr, (float*)d_out, MROWS, DOUT, DMOD);
}
